// GRU_72576357368187
// MI455X (gfx1250) — matrix operand order repaired and checked by reading, not compiled, not run on hardware
//
#include <hip/hip_runtime.h>
#include <hip/hip_bf16.h>

#define B_   32
#define P_   7
#define N_   207
#define T_   12
#define C_   256
#define HID_ 512
#define NT_  (N_ * T_)        // 2484
#define M_   (B_ * NT_)       // 79488
#define H3_  (3 * HID_)       // 1536

#define TM_  32               // M-tile rows per workgroup (2 row-blocks of 16)

// LDS row pitches (elements), padded by 8 bf16 (16 bytes) to break bank conflicts
#define XLD  264
#define HLD  520

typedef __attribute__((ext_vector_type(16))) __bf16 v16bf;
typedef __attribute__((ext_vector_type(8)))  float  v8f;

union FragU { v16bf v; uint4 q[2]; };

__device__ __forceinline__ unsigned short f2bf(float f) {
  unsigned u = __float_as_uint(f);
  u += 0x7FFFu + ((u >> 16) & 1u);        // round-to-nearest-even
  return (unsigned short)(u >> 16);
}

// A fragment (16x32 bf16, M x K): lane = (m, hf); q0 = A[m][k0+8*hf .. +7],
// q1 = A[m][k0+16+8*hf .. +7]  (ISA 7.12.2 16-bit A layout)
__device__ __forceinline__ v16bf ld_a(const unsigned short* tile, int ld, int k0) {
  const int lane = threadIdx.x & 31;
  const int m = lane & 15, hf = lane >> 4;
  const unsigned short* p0 = tile + m * ld + k0 + hf * 8;
  FragU f;
  f.q[0] = *(const uint4*)(p0);
  f.q[1] = *(const uint4*)(p0 + 16);
  return f.v;
}

// B fragment (32x16 bf16, K x N) from B^T stored row-major [N][Ktot]:
// lane = (n, hf) holds K = k0+8*hf .. +7 and k0+16+8*hf .. +7 of column n
__device__ __forceinline__ v16bf ld_b(const unsigned short* bt_col0, int ld, int k0) {
  const int lane = threadIdx.x & 31;
  const int n = lane & 15, hf = lane >> 4;
  const unsigned short* p0 = bt_col0 + (size_t)n * ld + k0 + hf * 8;
  FragU f;
  f.q[0] = *(const uint4*)(p0);
  f.q[1] = *(const uint4*)(p0 + 16);
  return f.v;
}

__device__ __forceinline__ v8f wmma_bf16(v16bf a, v16bf b, v8f c) {
  return __builtin_amdgcn_wmma_f32_16x16x32_bf16(false, a, false, b, (short)0, c,
                                                 false, false);
}

__device__ __forceinline__ float sigmoidf_(float x) {
  return 1.0f / (1.0f + __expf(-x));
}

__global__ __launch_bounds__(256) void gru_step(
    const float* __restrict__ X, const unsigned short* __restrict__ WxT,
    const unsigned short* __restrict__ WhT, const float* __restrict__ bx,
    const float* __restrict__ bh, float* __restrict__ H, int p) {
  __shared__ unsigned short sXbf[TM_ * XLD];   // X tile, bf16
  __shared__ unsigned short sHbf[TM_ * HLD];   // H tile, bf16 (GEMM operand)
  __shared__ unsigned short sRH [TM_ * HLD];   // R ⊙ H, bf16 (GEMM operand)
  __shared__ float          sHf [TM_ * HLD];   // H tile, f32 (update path)

  const int tid = threadIdx.x;
  const int r0  = blockIdx.x * TM_;

  // ---------- cooperative load: X tile [32x256], H tile [32x512] ----------
  {
    const int row = tid >> 3;          // 0..31
    const int l8  = tid & 7;           // 0..7
    const int m   = r0 + row;
    const size_t xbase =
        ((size_t)(m / NT_) * P_ + p) * (size_t)NT_ * C_ + (size_t)(m % NT_) * C_;
    const float4* xp = (const float4*)(X + xbase) + l8 * 8;
#pragma unroll
    for (int i = 0; i < 8; ++i) {
      float4 v = xp[i];
      int c = l8 * 32 + i * 4;
      sXbf[row * XLD + c + 0] = f2bf(v.x);
      sXbf[row * XLD + c + 1] = f2bf(v.y);
      sXbf[row * XLD + c + 2] = f2bf(v.z);
      sXbf[row * XLD + c + 3] = f2bf(v.w);
    }
    const float4* hp = (const float4*)(H + (size_t)m * HID_) + l8 * 16;
#pragma unroll
    for (int i = 0; i < 16; ++i) {
      float4 v = hp[i];
      int c = l8 * 64 + i * 4;
      sHf[row * HLD + c + 0] = v.x; sHbf[row * HLD + c + 0] = f2bf(v.x);
      sHf[row * HLD + c + 1] = v.y; sHbf[row * HLD + c + 1] = f2bf(v.y);
      sHf[row * HLD + c + 2] = v.z; sHbf[row * HLD + c + 2] = f2bf(v.z);
      sHf[row * HLD + c + 3] = v.w; sHbf[row * HLD + c + 3] = f2bf(v.w);
    }
  }
  __syncthreads();

  const int wid  = tid >> 5;     // wave 0..7, owns 64 H-columns
  const int lane = tid & 31;
  const int nl   = lane & 15;
  const int hf   = lane >> 4;

  const unsigned short* WxTp = WxT + (size_t)p * H3_ * C_;
  const unsigned short* WhTp = WhT + (size_t)p * H3_ * HID_;
  const float* bxp = bx + (size_t)p * H3_;
  const float* bhp = bh + (size_t)p * H3_;

  float zt[4][2][8];   // Z gate, kept in registers across the barrier
  float xh[4][2][8];   // x-part of candidate gate (+bx), kept in registers

  // ---------- phase 1: R and Z gates, x-part of H gate; write R⊙H ----------
#pragma unroll
  for (int t = 0; t < 4; ++t) {
    const int j = wid * 64 + t * 16;
    v8f accR[2] = {{}, {}};
    v8f accZ[2] = {{}, {}};
    v8f accH[2] = {{}, {}};
#pragma unroll
    for (int k0 = 0; k0 < C_; k0 += 32) {          // X @ Wx, all three gates
      v16bf a0 = ld_a(sXbf,            XLD, k0);   // rows 0..15
      v16bf a1 = ld_a(sXbf + 16 * XLD, XLD, k0);   // rows 16..31
      v16bf br = ld_b(WxTp + (size_t)(j       ) * C_, C_, k0);
      v16bf bz = ld_b(WxTp + (size_t)(512 + j ) * C_, C_, k0);
      v16bf bc = ld_b(WxTp + (size_t)(1024 + j) * C_, C_, k0);
      accR[0] = wmma_bf16(a0, br, accR[0]); accR[1] = wmma_bf16(a1, br, accR[1]);
      accZ[0] = wmma_bf16(a0, bz, accZ[0]); accZ[1] = wmma_bf16(a1, bz, accZ[1]);
      accH[0] = wmma_bf16(a0, bc, accH[0]); accH[1] = wmma_bf16(a1, bc, accH[1]);
    }
#pragma unroll
    for (int k0 = 0; k0 < HID_; k0 += 32) {        // H @ Wh for r,z
      v16bf a0 = ld_a(sHbf,            HLD, k0);
      v16bf a1 = ld_a(sHbf + 16 * HLD, HLD, k0);
      v16bf br = ld_b(WhTp + (size_t)(j      ) * HID_, HID_, k0);
      v16bf bz = ld_b(WhTp + (size_t)(512 + j) * HID_, HID_, k0);
      accR[0] = wmma_bf16(a0, br, accR[0]); accR[1] = wmma_bf16(a1, br, accR[1]);
      accZ[0] = wmma_bf16(a0, bz, accZ[0]); accZ[1] = wmma_bf16(a1, bz, accZ[1]);
    }
    const float bR = bxp[j + nl]        + bhp[j + nl];
    const float bZ = bxp[512 + j + nl]  + bhp[512 + j + nl];
    const float bX = bxp[1024 + j + nl];
#pragma unroll
    for (int rb = 0; rb < 2; ++rb) {
#pragma unroll
      for (int r2 = 0; r2 < 8; ++r2) {
        const int mm = rb * 16 + r2 + hf * 8;      // C/D layout: M = vgpr + 8*half
        float rt       = sigmoidf_(accR[rb][r2] + bR);
        zt[t][rb][r2]  = sigmoidf_(accZ[rb][r2] + bZ);
        xh[t][rb][r2]  = accH[rb][r2] + bX;
        float hv       = sHf[mm * HLD + j + nl];
        sRH[mm * HLD + j + nl] = f2bf(rt * hv);
      }
    }
  }
  __syncthreads();

  // ---------- phase 2: hh = (R⊙H) @ Wh3 ; candidate; state update ----------
#pragma unroll
  for (int t = 0; t < 4; ++t) {
    const int j = wid * 64 + t * 16;
    v8f accC[2] = {{}, {}};
#pragma unroll
    for (int k0 = 0; k0 < HID_; k0 += 32) {
      v16bf a0 = ld_a(sRH,            HLD, k0);
      v16bf a1 = ld_a(sRH + 16 * HLD, HLD, k0);
      v16bf bc = ld_b(WhTp + (size_t)(1024 + j) * HID_, HID_, k0);
      accC[0] = wmma_bf16(a0, bc, accC[0]);
      accC[1] = wmma_bf16(a1, bc, accC[1]);
    }
    const float bH = bhp[1024 + j + nl];
#pragma unroll
    for (int rb = 0; rb < 2; ++rb) {
#pragma unroll
      for (int r2 = 0; r2 < 8; ++r2) {
        const int mm   = rb * 16 + r2 + hf * 8;
        float cand = tanhf(xh[t][rb][r2] + accC[rb][r2] + bH);
        float hold = sHf[mm * HLD + j + nl];
        float z    = zt[t][rb][r2];
        H[(size_t)(r0 + mm) * HID_ + j + nl] = z * hold + (1.0f - z) * cand;
      }
    }
  }
}

// ---------- one-time prep: weight transpose + f32->bf16 ----------
__global__ void wx_transpose(const float* __restrict__ Wx,
                             unsigned short* __restrict__ WxT) {
  const int j  = blockIdx.x;   // 0..1535 (gate column -> WxT row)
  const int pp = blockIdx.y;   // period
  for (int c = threadIdx.x; c < C_; c += blockDim.x)
    WxT[((size_t)pp * H3_ + j) * C_ + c] =
        f2bf(Wx[((size_t)pp * C_ + c) * H3_ + j]);
}

__global__ void wh_transpose(const float* __restrict__ Wh,
                             unsigned short* __restrict__ WhT) {
  const int j  = blockIdx.x;
  const int pp = blockIdx.y;
  for (int h = threadIdx.x; h < HID_; h += blockDim.x)
    WhT[((size_t)pp * H3_ + j) * HID_ + h] =
        f2bf(Wh[((size_t)pp * HID_ + h) * H3_ + j]);
}

__global__ void zero_f32(float* __restrict__ o, int n) {
  int i = blockIdx.x * blockDim.x + threadIdx.x;
  if (i < n) o[i] = 0.0f;
}

extern "C" void kernel_launch(void* const* d_in, const int* in_sizes, int n_in,
                              void* d_out, int out_size, void* d_ws, size_t ws_size,
                              hipStream_t stream) {
  const float* X  = (const float*)d_in[0];
  const float* Wx = (const float*)d_in[1];
  const float* bx = (const float*)d_in[2];
  const float* Wh = (const float*)d_in[3];
  const float* bh = (const float*)d_in[4];
  float* H = (float*)d_out;   // H state lives in d_out ([B,N,T,H] == output)

  unsigned short* WxT = (unsigned short*)d_ws;
  unsigned short* WhT =
      (unsigned short*)((char*)d_ws + (size_t)P_ * H3_ * C_ * sizeof(unsigned short));

  const int n = M_ * HID_;
  zero_f32<<<(n + 255) / 256, 256, 0, stream>>>(H, n);          // H0 = 0
  wx_transpose<<<dim3(H3_, P_), 256, 0, stream>>>(Wx, WxT);
  wh_transpose<<<dim3(H3_, P_), 256, 0, stream>>>(Wh, WhT);

  for (int p = 0; p < P_; ++p)
    gru_step<<<M_ / TM_, 256, 0, stream>>>(X, WxT, WhT, bx, bh, H, p);
}
